// FBSDE_22548578304355
// MI455X (gfx1250) — compile-verified, matrix operand order checked by reading, not run on hardware
//
#include <hip/hip_runtime.h>
#include <hip/hip_bf16.h>
#include <math.h>

typedef __attribute__((ext_vector_type(16))) __bf16 v16bf;
typedef __attribute__((ext_vector_type(8)))  __bf16 v8bf;
typedef __attribute__((ext_vector_type(8)))  float  v8f;

#define F_MU 0.5f
// B=512, N=65, D=64, H=256, M = B*N = 33280

// ---------------------------------------------------------------------------
// WMMA helpers: bf16 16x16x32, f32 accumulate
// ---------------------------------------------------------------------------
__device__ __forceinline__ v8f wmma_bf16(v16bf a, v16bf b, v8f c) {
  return __builtin_amdgcn_wmma_f32_16x16x32_bf16(
      /*neg_a=*/false, a, /*neg_b=*/false, b,
      /*c_mod=*/(short)0, c, /*reuse_a=*/false, /*reuse_b=*/false);
}

// Load a 16x32 bf16 fragment from row-major storage (A from activations,
// B from pre-transposed weights Wt[N][Kpad]).  ISA layout (CDNA5 ISA):
// lane<16: row=lane, K 0..7 in halves 0..7, K 16..23 in halves 8..15;
// lane>=16: row=lane-16, K 8..15 / 24..31.  -> two aligned 16B loads/lane.
__device__ __forceinline__ v16bf load_frag(const __bf16* base, int ld,
                                           int r16, int hi) {
  const __bf16* p = base + r16 * ld + hi * 8;
  v8bf lo = *(const v8bf*)(p);
  v8bf hh = *(const v8bf*)(p + 16);
  v16bf o;
#pragma unroll
  for (int i = 0; i < 8; ++i) { o[i] = lo[i]; o[i + 8] = hh[i]; }
  return o;
}

// ---------------------------------------------------------------------------
// Weight prep (single launch): f32 W[K][N] -> bf16 Wt[N][Kpad], zero padded
// ---------------------------------------------------------------------------
struct WPrep {
  const float* src[9];
  __bf16*      dst[9];
  int K[9], N[9], Kp[9];
};

__global__ void prep_weights(WPrep p) {
  for (int s = 0; s < 9; ++s) {
    const float* W = p.src[s];
    __bf16* Wt = p.dst[s];
    int K = p.K[s], Nn = p.N[s], Kp = p.Kp[s];
    int tot = Nn * Kp;
    for (int i = blockIdx.x * blockDim.x + threadIdx.x; i < tot;
         i += gridDim.x * blockDim.x) {
      int n = i / Kp, k = i % Kp;
      Wt[i] = (k < K) ? (__bf16)W[(size_t)k * Nn + n] : (__bf16)0.f;
    }
  }
}

// ---------------------------------------------------------------------------
// Persistent Euler rollout: ONE launch, time loop inside.  Each block owns 16
// batch rows for all 64 steps (row-independent recursion => no cross-block
// sync).  x lives in LDS; only xpath log-writes / noise reads touch HBM.
// grid 32 x 256 threads.
// ---------------------------------------------------------------------------
__global__ void sde_rollout(const float* __restrict__ ts, const float* __restrict__ x0,
                            const float* __restrict__ noise,
                            const __bf16* __restrict__ aw1t, const float* __restrict__ ab1,
                            const __bf16* __restrict__ aw2t, const float* __restrict__ ab2,
                            const __bf16* __restrict__ aw3t, const float* __restrict__ ab3,
                            float* __restrict__ xpath) {
  __shared__ __align__(16) __bf16 sTX[16 * 96];
  __shared__ __align__(16) __bf16 sH1[16 * 256];
  __shared__ __align__(16) __bf16 sH2[16 * 256];
  __shared__ float sPart[2 * 16 * 64];   // split-K partials for layer 3
  __shared__ float sX[16 * 64];          // current x, persistent across steps
  int tid = threadIdx.x;
  int rowbase = blockIdx.x * 16;

  for (int i = tid; i < 16 * 64; i += 256) {
    int r = i >> 6, c = i & 63;
    float v = x0[(size_t)(rowbase + r) * 64 + c];
    sX[i] = v;
    xpath[((size_t)(rowbase + r) * 65) * 64 + c] = v;   // x at n=0
  }
  __syncthreads();

  int wave = tid >> 5, lane = tid & 31, r16 = lane & 15, hi = lane >> 4;

  for (int n = 0; n < 64; ++n) {
    float tnext = ts[n + 1], hstep = tnext - ts[n];
    float sqh = sqrtf(hstep);

    // stage tx = [t, x_n, 0-pad to 96] as bf16 (from LDS x)
    for (int i = tid; i < 16 * 96; i += 256) {
      int r = i / 96, c = i % 96;
      float v = 0.f;
      if (c == 0) v = tnext;
      else if (c < 65) v = sX[(r << 6) + (c - 1)];
      sTX[i] = (__bf16)v;
    }
    __syncthreads();

    // L1: h1 = relu(tx @ w1 + b1)   (16x256, K=96) - 16 tiles, 2/wave
    for (int t = wave; t < 16; t += 8) {
      v8f acc = {};
      for (int k = 0; k < 96; k += 32) {
        v16bf af = load_frag(sTX + k, 96, r16, hi);
        v16bf bf = load_frag(aw1t + (size_t)(t * 16) * 96 + k, 96, r16, hi);
        acc = wmma_bf16(af, bf, acc);
      }
      int col = t * 16 + r16;
      float bv = ab1[col];
#pragma unroll
      for (int e = 0; e < 8; ++e) {
        int row = e + 8 * hi;
        float v = acc[e] + bv;
        v = v > 0.f ? v : 0.f;
        sH1[row * 256 + col] = (__bf16)v;
      }
    }
    __syncthreads();

    // L2: h2 = relu(h1 @ w2 + b2)   (16x256, K=256)
    for (int t = wave; t < 16; t += 8) {
      v8f acc = {};
      for (int k = 0; k < 256; k += 32) {
        v16bf af = load_frag(sH1 + k, 256, r16, hi);
        v16bf bf = load_frag(aw2t + (size_t)(t * 16) * 256 + k, 256, r16, hi);
        acc = wmma_bf16(af, bf, acc);
      }
      int col = t * 16 + r16;
      float bv = ab2[col];
#pragma unroll
      for (int e = 0; e < 8; ++e) {
        int row = e + 8 * hi;
        float v = acc[e] + bv;
        v = v > 0.f ? v : 0.f;
        sH2[row * 256 + col] = (__bf16)v;
      }
    }
    __syncthreads();

    // L3: a = h2 @ w3   (16x64, K=256) split-K: 4 col-tiles x 2 K-halves
    {
      int ct = wave & 3, kh = wave >> 2;
      v8f acc = {};
      for (int k = kh * 128; k < kh * 128 + 128; k += 32) {
        v16bf af = load_frag(sH2 + k, 256, r16, hi);
        v16bf bf = load_frag(aw3t + (size_t)(ct * 16) * 256 + k, 256, r16, hi);
        acc = wmma_bf16(af, bf, acc);
      }
      int col = ct * 16 + r16;
#pragma unroll
      for (int e = 0; e < 8; ++e) {
        int row = e + 8 * hi;
        sPart[(kh * 16 + row) * 64 + col] = acc[e];
      }
    }
    __syncthreads();

    // Euler update (elementwise): x <- x + (MU*x + a)*h + dW
    for (int i = tid; i < 16 * 64; i += 256) {
      int r = i >> 6, c = i & 63;
      float a = sPart[r * 64 + c] + sPart[(16 + r) * 64 + c] + ab3[c];
      float x = sX[i];
      int b = rowbase + r;
      float dw = noise[((size_t)b * 65 + n) * 64 + c] * sqh;
      float xn = x + (F_MU * x + a) * hstep + dw;     // SIGMA = 1
      sX[i] = xn;
      xpath[((size_t)b * 65 + n + 1) * 64 + c] = xn;
    }
    __syncthreads();
  }
}

// ---------------------------------------------------------------------------
// Build padded bf16 [t, x] input for Y/Z nets, all M = 33280 rows
// ---------------------------------------------------------------------------
__global__ void build_txall(const float* __restrict__ ts, const float* __restrict__ xpath,
                            __bf16* __restrict__ txall) {
  const int TOT = 33280 * 96;
  for (int i = blockIdx.x * blockDim.x + threadIdx.x; i < TOT;
       i += gridDim.x * blockDim.x) {
    int m = i / 96, c = i % 96;
    float v = 0.f;
    if (c == 0) v = ts[m % 65];
    else if (c < 65) v = xpath[(size_t)m * 64 + (c - 1)];
    txall[i] = (__bf16)v;
  }
}

// ---------------------------------------------------------------------------
// Generic FFN layer: out[M][Ncols] = act(A[M][Kpad] @ Wt^T + bias)
// grid.x = M/32, 256 threads
// ---------------------------------------------------------------------------
template <bool RELU, bool OUTBF>
__global__ void ffn_layer(const __bf16* __restrict__ A, int Kpad,
                          const __bf16* __restrict__ Wt, const float* __restrict__ bias,
                          int Ncols, void* __restrict__ out) {
  __shared__ __align__(16) __bf16 Alds[32 * 256];
  int tid = threadIdx.x;
  const __bf16* Ablk = A + (size_t)blockIdx.x * 32 * Kpad;
  for (int i = tid; i < (32 * Kpad) / 8; i += 256)
    ((v8bf*)Alds)[i] = ((const v8bf*)Ablk)[i];
  __syncthreads();

  int wave = tid >> 5, lane = tid & 31, r16 = lane & 15, hi = lane >> 4;
  int ctiles = Ncols >> 4;
  for (int t = wave; t < 2 * ctiles; t += 8) {
    int mt = t / ctiles, ct = t % ctiles;
    v8f acc = {};
    for (int k = 0; k < Kpad; k += 32) {
      v16bf af = load_frag(Alds + (mt * 16) * Kpad + k, Kpad, r16, hi);
      v16bf bf = load_frag(Wt + (size_t)(ct * 16) * Kpad + k, Kpad, r16, hi);
      acc = wmma_bf16(af, bf, acc);
    }
    int col = ct * 16 + r16;
    float bv = bias[col];
#pragma unroll
    for (int e = 0; e < 8; ++e) {
      int row = mt * 16 + e + 8 * hi;
      size_t gidx = ((size_t)blockIdx.x * 32 + row) * Ncols + col;
      float v = acc[e] + bv;
      if (RELU) v = v > 0.f ? v : 0.f;
      if (OUTBF) ((__bf16*)out)[gidx] = (__bf16)v;
      else       ((float*)out)[gidx] = v;
    }
  }
}

// ---------------------------------------------------------------------------
// Fused Z final layer (256 -> 4096) + stoch = Z . dW  (never materialize Z)
// Wave w: mt = w>>2 (row half), il = w&3 (Z-row i within 256-col chunk).
// grid.x = 1040, 256 threads
// ---------------------------------------------------------------------------
__global__ void z_last_stoch(const __bf16* __restrict__ H,
                             const __bf16* __restrict__ Wt /*4096x256*/,
                             const float* __restrict__ zb3,
                             const float* __restrict__ noise,
                             const float* __restrict__ ts,
                             float* __restrict__ stoch) {
  __shared__ __align__(16) __bf16 Alds[32 * 256];
  __shared__ float dwl[32 * 64];
  int tid = threadIdx.x;
  const __bf16* Hb = H + (size_t)blockIdx.x * 32 * 256;
  for (int i = tid; i < 32 * 256 / 8; i += 256)
    ((v8bf*)Alds)[i] = ((const v8bf*)Hb)[i];
  for (int i = tid; i < 32 * 64; i += 256) {
    int r = i >> 6, j = i & 63;
    int m = blockIdx.x * 32 + r;
    int n = m % 65;
    float v = 0.f;
    if (n < 64) v = noise[(size_t)m * 64 + j] * sqrtf(ts[n + 1] - ts[n]);
    dwl[i] = v;
  }
  __syncthreads();

  int wave = tid >> 5, lane = tid & 31, r16 = lane & 15, hi = lane >> 4;
  int mt = wave >> 2, il = wave & 3;
  for (int chunk = 0; chunk < 16; ++chunk) {
    int i = chunk * 4 + il;               // Z row index 0..63
    float sacc[8];
#pragma unroll
    for (int e = 0; e < 8; ++e) sacc[e] = 0.f;
    for (int ct = 0; ct < 4; ++ct) {      // 4 tiles = all 64 j for this i
      v8f acc = {};
      const __bf16* Bbase = Wt + (size_t)(i * 64 + ct * 16) * 256;
      for (int k = 0; k < 256; k += 32) {
        v16bf af = load_frag(Alds + (mt * 16) * 256 + k, 256, r16, hi);
        v16bf bf = load_frag(Bbase + k, 256, r16, hi);
        acc = wmma_bf16(af, bf, acc);
      }
      int j = ct * 16 + r16;
      float bv = zb3[i * 64 + j];
#pragma unroll
      for (int e = 0; e < 8; ++e) {
        int row = mt * 16 + e + 8 * hi;
        sacc[e] += (acc[e] + bv) * dwl[row * 64 + j];
      }
    }
    // reduce over the 16 j-lanes (lanes 0..15 and 16..31 independently)
#pragma unroll
    for (int e = 0; e < 8; ++e) {
      float v = sacc[e];
      v += __shfl_xor(v, 1);
      v += __shfl_xor(v, 2);
      v += __shfl_xor(v, 4);
      v += __shfl_xor(v, 8);
      sacc[e] = v;
    }
    if (r16 == 0) {
#pragma unroll
      for (int e = 0; e < 8; ++e) {
        int m = blockIdx.x * 32 + mt * 16 + e + 8 * hi;
        stoch[(size_t)m * 64 + i] = sacc[e];
      }
    }
  }
}

// ---------------------------------------------------------------------------
// Loss: sum_n mean_{b,d}(pred-target)^2 + mean_{b,d}(Y_T - x_T)^2
// ---------------------------------------------------------------------------
__global__ void zero_out(float* out) { out[0] = 0.f; }

__global__ void loss_kernel(const float* __restrict__ Y, const float* __restrict__ xpath,
                            const float* __restrict__ stoch, const float* __restrict__ ts,
                            float* __restrict__ out) {
  __shared__ float red[256];
  float acc = 0.f;
  int stride = gridDim.x * blockDim.x;
  const int TOT = 512 * 64 * 64;
  for (int i = blockIdx.x * blockDim.x + threadIdx.x; i < TOT; i += stride) {
    int d = i & 63;
    int n = (i >> 6) & 63;
    int b = i >> 12;
    size_t m = (size_t)b * 65 + n;
    float h = ts[n + 1] - ts[n];
    float y = Y[m * 64 + d];
    float pred = y - (F_MU * y + xpath[m * 64 + d]) * h + stoch[m * 64 + d];
    float diff = pred - Y[(m + 1) * 64 + d];
    acc += diff * diff;
  }
  const int TOT2 = 512 * 64;
  for (int i = blockIdx.x * blockDim.x + threadIdx.x; i < TOT2; i += stride) {
    int d = i & 63, b = i >> 6;
    size_t m = (size_t)b * 65 + 64;
    float diff = Y[m * 64 + d] - xpath[m * 64 + d];
    acc += diff * diff;
  }
  acc *= (1.0f / (512.0f * 64.0f));
  red[threadIdx.x] = acc;
  __syncthreads();
  for (int s = 128; s > 0; s >>= 1) {
    if (threadIdx.x < s) red[threadIdx.x] += red[threadIdx.x + s];
    __syncthreads();
  }
  if (threadIdx.x == 0) atomicAdd(out, red[0]);
}

// ---------------------------------------------------------------------------
extern "C" void kernel_launch(void* const* d_in, const int* in_sizes, int n_in,
                              void* d_out, int out_size, void* d_ws, size_t ws_size,
                              hipStream_t stream) {
  (void)in_sizes; (void)n_in; (void)out_size; (void)ws_size;
  const float* aw1 = (const float*)d_in[0];  const float* ab1 = (const float*)d_in[1];
  const float* aw2 = (const float*)d_in[2];  const float* ab2 = (const float*)d_in[3];
  const float* aw3 = (const float*)d_in[4];  const float* ab3 = (const float*)d_in[5];
  const float* yw1 = (const float*)d_in[6];  const float* yb1 = (const float*)d_in[7];
  const float* yw2 = (const float*)d_in[8];  const float* yb2 = (const float*)d_in[9];
  const float* yw3 = (const float*)d_in[10]; const float* yb3 = (const float*)d_in[11];
  const float* zw1 = (const float*)d_in[12]; const float* zb1 = (const float*)d_in[13];
  const float* zw2 = (const float*)d_in[14]; const float* zb2 = (const float*)d_in[15];
  const float* zw3 = (const float*)d_in[16]; const float* zb3 = (const float*)d_in[17];
  const float* ts  = (const float*)d_in[18];
  const float* x0  = (const float*)d_in[19];
  const float* noi = (const float*)d_in[20];
  float* out = (float*)d_out;

  // workspace carve-up
  char* ws = (char*)d_ws;
  size_t o = 0;
  auto carve = [&](size_t bytes) { char* p = ws + o; o = (o + bytes + 255) & ~(size_t)255; return p; };
  __bf16* aw1t = (__bf16*)carve(256 * 96 * 2);
  __bf16* aw2t = (__bf16*)carve(256 * 256 * 2);
  __bf16* aw3t = (__bf16*)carve(64 * 256 * 2);
  __bf16* yw1t = (__bf16*)carve(256 * 96 * 2);
  __bf16* yw2t = (__bf16*)carve(256 * 256 * 2);
  __bf16* yw3t = (__bf16*)carve(64 * 256 * 2);
  __bf16* zw1t = (__bf16*)carve(256 * 96 * 2);
  __bf16* zw2t = (__bf16*)carve(256 * 256 * 2);
  __bf16* zw3t = (__bf16*)carve((size_t)4096 * 256 * 2);
  float*  xpath = (float*)carve((size_t)512 * 65 * 64 * 4);
  __bf16* txall = (__bf16*)carve((size_t)33280 * 96 * 2);
  __bf16* hA    = (__bf16*)carve((size_t)33280 * 256 * 2);
  __bf16* hB    = (__bf16*)carve((size_t)33280 * 256 * 2);
  float*  Ybuf  = (float*)carve((size_t)33280 * 64 * 4);
  float*  stoch = (float*)carve((size_t)33280 * 64 * 4);

  // single-launch weight prep
  WPrep p;
  p.src[0] = aw1; p.dst[0] = aw1t; p.K[0] = 65;  p.N[0] = 256;  p.Kp[0] = 96;
  p.src[1] = aw2; p.dst[1] = aw2t; p.K[1] = 256; p.N[1] = 256;  p.Kp[1] = 256;
  p.src[2] = aw3; p.dst[2] = aw3t; p.K[2] = 256; p.N[2] = 64;   p.Kp[2] = 256;
  p.src[3] = yw1; p.dst[3] = yw1t; p.K[3] = 65;  p.N[3] = 256;  p.Kp[3] = 96;
  p.src[4] = yw2; p.dst[4] = yw2t; p.K[4] = 256; p.N[4] = 256;  p.Kp[4] = 256;
  p.src[5] = yw3; p.dst[5] = yw3t; p.K[5] = 256; p.N[5] = 64;   p.Kp[5] = 256;
  p.src[6] = zw1; p.dst[6] = zw1t; p.K[6] = 65;  p.N[6] = 256;  p.Kp[6] = 96;
  p.src[7] = zw2; p.dst[7] = zw2t; p.K[7] = 256; p.N[7] = 256;  p.Kp[7] = 256;
  p.src[8] = zw3; p.dst[8] = zw3t; p.K[8] = 256; p.N[8] = 4096; p.Kp[8] = 256;
  prep_weights<<<2048, 256, 0, stream>>>(p);

  // single-launch persistent Euler rollout (blocks own disjoint batch rows)
  sde_rollout<<<32, 256, 0, stream>>>(ts, x0, noi, aw1t, ab1, aw2t, ab2, aw3t, ab3, xpath);

  build_txall<<<4096, 256, 0, stream>>>(ts, xpath, txall);

  // Y network
  ffn_layer<true,  true ><<<1040, 256, 0, stream>>>(txall, 96,  yw1t, yb1, 256, hA);
  ffn_layer<true,  true ><<<1040, 256, 0, stream>>>(hA,    256, yw2t, yb2, 256, hB);
  ffn_layer<false, false><<<1040, 256, 0, stream>>>(hB,    256, yw3t, yb3, 64,  Ybuf);
  // Z network hidden layers (reuse hA/hB)
  ffn_layer<true,  true ><<<1040, 256, 0, stream>>>(txall, 96,  zw1t, zb1, 256, hA);
  ffn_layer<true,  true ><<<1040, 256, 0, stream>>>(hA,    256, zw2t, zb2, 256, hB);
  // fused Z final layer + dW contraction
  z_last_stoch<<<1040, 256, 0, stream>>>(hB, zw3t, zb3, noi, ts, stoch);

  zero_out<<<1, 1, 0, stream>>>(out);
  loss_kernel<<<256, 256, 0, stream>>>(Ybuf, xpath, stoch, ts, out);
}
